// Voxel_RefinerXL_78520592106212
// MI455X (gfx1250) — matrix operand order repaired and checked by reading, not compiled, not run on hardware
//
#include <hip/hip_runtime.h>
#include <hip/hip_bf16.h>

// ---------------------------------------------------------------------------
// Voxel_RefinerXL refinement pipeline for MI455X (gfx1250, wave32, WMMA).
//
// Conv stages = implicit GEMM on v_wmma_f32_16x16x32_bf16:
//   * GEMM K = tap*8 + ic, LDS tile channel-innermost, so each lane's B
//     operand per 32-K chunk is two contiguous aligned ds_load_b128
//   * tap offsets are constexpr (selected per K-half with one cndmask)
//   * A operand pre-packed once per layer into global scratch (lane layout
//     with contiguous K = 16*half + e), fetched as clause'd global_load_b128 pairs
// Adaptive-conv applications are memory-bound VALU kernels.
// ---------------------------------------------------------------------------

#define DHW  128
#define NVOX (128 * 128 * 128)

typedef __attribute__((ext_vector_type(16))) __bf16 v16bf;
typedef __attribute__((ext_vector_type(8)))  __bf16 v8bf;
typedef __attribute__((ext_vector_type(8)))  float  v8f;

__device__ __forceinline__ unsigned short f_to_bfu(float f) {
    unsigned u = __builtin_bit_cast(unsigned, f);
    unsigned r = u + 0x7FFFu + ((u >> 16) & 1u);   // round-to-nearest-even
    return (unsigned short)(r >> 16);
}

// tap -> offset (u16 units) inside xs[z][y][x][ic]; tap 27 = zero pad
__device__ constexpr int tapOff(int t) {
    return (t < 27) ? (((t / 9) * 6 + ((t % 9) / 3)) * 18 + (t % 3)) * 8 : 0;
}

#define MODE_RELU 0
#define MODE_BIAS 1
#define MODE_NORM 2
#define MODE_OC1  3

// ---------------------------------------------------------------------------
// One-shot A-operand pack: Wg[OC][216] (fp32, OIDHW order) -> bf16 lane layout
// Apk[(t*7+kc)*32+ln][e], with GEMM K = tap*8 + ic.
// ---------------------------------------------------------------------------
template <int OCT, int OC>
__global__ __launch_bounds__(256)
void pack_wmma_A(const float* __restrict__ Wg, unsigned short* __restrict__ Apk)
{
    int i = blockIdx.x * 256 + threadIdx.x;
    if (i >= OCT * 7 * 32 * 16) return;
    int e = i & 15, ln = (i >> 4) & 31, k2 = i >> 9;
    int kc = k2 % 7, t = k2 / 7;
    // A-operand K per (element, lane-half): contiguous, K = kc*32 + 16*half + e (matches B)
    int K  = kc * 32 + e + ((ln & 16) ? 16 : 0);
    int oc = t * 16 + (ln & 15);
    float v = 0.f;
    if (oc < OC && K < 216) {
        int tap = K >> 3, ic = K & 7;            // K = tap*8 + ic
        v = Wg[oc * 216 + ic * 27 + tap];
    }
    Apk[i] = f_to_bfu(v);
}

// ---------------------------------------------------------------------------
// WMMA 3x3x3 conv, 8 input channels, OC output channels.
//   GEMM: M = OC (16*OCT rows), K = 216 pad 224, N = 16 voxels per wave.
//   Workgroup tile = 2(D) x 4(H) x 16(W), 8 waves.
// ---------------------------------------------------------------------------
template <int OCT, int MODE>
__global__ __launch_bounds__(256)
void conv3x3x3_wmma(const float* __restrict__ X,            // [8][NVOX]
                    const unsigned short* __restrict__ Apk, // packed A (bf16)
                    const float* __restrict__ bias,         // [OC] or null
                    float* __restrict__ Y)                  // [OC][NVOX]
{
    // LDS input tile, channel-innermost: xs[z 0..3][y 0..5][x 0..17][ic 0..7]
    __shared__ __align__(16) unsigned short xs[4 * 6 * 18 * 8];

    const int tid = threadIdx.x;
    const int db = blockIdx.z * 2;   // D tile base
    const int hb = blockIdx.y * 4;   // H tile base
    const int wb = blockIdx.x * 16;  // W tile base

    // ---- stage input tile with halo (fp32 -> bf16), coalesced over x ----
    for (int i = tid; i < 8 * 4 * 6 * 18; i += 256) {
        int ic = i / 432, r = i % 432;
        int zz = r / 108, r2 = r % 108, yy = r2 / 18, xw = r2 % 18;
        int gd = db + zz - 1, gh = hb + yy - 1, gw = wb + xw - 1;
        float v = 0.f;
        if ((unsigned)gd < 128u && (unsigned)gh < 128u && (unsigned)gw < 128u)
            v = X[ic * NVOX + (gd * 128 + gh) * 128 + gw];
        xs[((zz * 6 + yy) * 18 + xw) * 8 + ic] = f_to_bfu(v);
    }
    __syncthreads();

    // ---- per-wave GEMM: 16(M) x 16(N=voxels) x 224(K) ----
    const int wv = tid >> 5, ln = tid & 31;
    const int dz = wv >> 2, hy = wv & 3;          // wave's (d,h) within tile
    const int n  = ln & 15;                       // voxel column (N)
    const bool khalf = (ln & 16) != 0;            // which K-half this lane holds
    const int bb = ((dz * 6 + hy) * 18 + n) * 8;  // lane base in xs (u16 units)

    v8f acc0 = {0.f, 0.f, 0.f, 0.f, 0.f, 0.f, 0.f, 0.f};
    v8f acc1 = {0.f, 0.f, 0.f, 0.f, 0.f, 0.f, 0.f, 0.f};

#pragma unroll
    for (int kc = 0; kc < 7; kc++) {
        // taps for this chunk: lo half -> 4kc,4kc+1 ; hi half -> 4kc+2,4kc+3
        int o0 = khalf ? tapOff(kc * 4 + 2) : tapOff(kc * 4 + 0);
        int o1 = khalf ? tapOff(kc * 4 + 3) : tapOff(kc * 4 + 1);
        v8bf blo = *reinterpret_cast<const v8bf*>(&xs[bb + o0]);
        v8bf bhi = *reinterpret_cast<const v8bf*>(&xs[bb + o1]);
        v16bf b = __builtin_shufflevector(blo, bhi,
            0, 1, 2, 3, 4, 5, 6, 7, 8, 9, 10, 11, 12, 13, 14, 15);
        v16bf a0 = *reinterpret_cast<const v16bf*>(Apk + (kc * 32 + ln) * 16);
        acc0 = __builtin_amdgcn_wmma_f32_16x16x32_bf16(
            false, a0, false, b, (short)0, acc0, false, false);
        if (OCT == 2) {
            v16bf a1 = *reinterpret_cast<const v16bf*>(Apk + ((7 + kc) * 32 + ln) * 16);
            acc1 = __builtin_amdgcn_wmma_f32_16x16x32_bf16(
                false, a1, false, b, (short)0, acc1, false, false);
        }
    }

    // ---- epilogue: C/D layout = lane<16: M=r, N=lane ; lane>=16: M=r+8 ----
    const int mofs = khalf ? 8 : 0;
    const int vox  = ((db + dz) * 128 + (hb + hy)) * 128 + (wb + n);

    if (MODE == MODE_NORM) {
        float s = 0.f;
#pragma unroll
        for (int r = 0; r < 8; r++) s += fabsf(acc0[r]);
#pragma unroll
        for (int r = 0; r < 8; r++) {
            int oc = 16 + r + mofs;
            if (oc < 27) s += fabsf(acc1[r]);
        }
        s += __shfl_xor(s, 16, 32);              // combine lane n with n+16
        float inv = 1.f / fmaxf(s, 1e-12f);
#pragma unroll
        for (int r = 0; r < 8; r++)
            Y[(r + mofs) * NVOX + vox] = acc0[r] * inv;
#pragma unroll
        for (int r = 0; r < 8; r++) {
            int oc = 16 + r + mofs;
            if (oc < 27) Y[oc * NVOX + vox] = acc1[r] * inv;
        }
    } else if (MODE == MODE_OC1) {
        if (!khalf) Y[vox] = acc0[0] + bias[0];
    } else {
#pragma unroll
        for (int r = 0; r < 8; r++) {
            int oc = r + mofs;
            if (oc < 8) {
                float v = acc0[r] + bias[oc];
                if (MODE == MODE_RELU) v = fmaxf(v, 0.f);
                Y[oc * NVOX + vox] = v;
            }
        }
    }
}

// ---------------------------------------------------------------------------
// Adaptive conv: per-voxel 27-tap kernel shared across C channels (mem-bound)
// ---------------------------------------------------------------------------
template <int C>
__global__ __launch_bounds__(256)
void adaptive_conv_k(const float* __restrict__ X,   // [C][NVOX]
                     const float* __restrict__ Wv,  // [27][NVOX] (L1-normalized)
                     float* __restrict__ Y)         // [C][NVOX]
{
    int v = blockIdx.x * 256 + threadIdx.x;
    int xw = v & 127, hy = (v >> 7) & 127, dz = v >> 14;
    float acc[C];
#pragma unroll
    for (int c = 0; c < C; c++) acc[c] = 0.f;

#pragma unroll
    for (int t = 0; t < 27; t++) {
        const int tz = t / 9 - 1, ty = (t % 9) / 3 - 1, tx = t % 3 - 1;
        int dd = dz + tz, hh = hy + ty, ww = xw + tx;
        float wv = Wv[t * NVOX + v];
        if ((unsigned)dd < 128u && (unsigned)hh < 128u && (unsigned)ww < 128u) {
            int sv = (dd * 128 + hh) * 128 + ww;
#pragma unroll
            for (int c = 0; c < C; c++)
                acc[c] = fmaf(wv, X[c * NVOX + sv], acc[c]);
        }
    }
#pragma unroll
    for (int c = 0; c < C; c++) Y[c * NVOX + v] = acc[c];
}

// ---------------------------------------------------------------------------
extern "C" void kernel_launch(void* const* d_in, const int* in_sizes, int n_in,
                              void* d_out, int out_size, void* d_ws, size_t ws_size,
                              hipStream_t stream) {
    (void)in_sizes; (void)n_in; (void)out_size; (void)ws_size;

    const float* x      = (const float*)d_in[0];
    const float* ac1_w1 = (const float*)d_in[1];
    const float* ac1_b1 = (const float*)d_in[2];
    const float* ac1_w2 = (const float*)d_in[3];
    const float* ac2_w1 = (const float*)d_in[4];
    const float* ac2_b1 = (const float*)d_in[5];
    const float* ac2_w2 = (const float*)d_in[6];
    const float* ac3_w1 = (const float*)d_in[7];
    const float* ac3_b1 = (const float*)d_in[8];
    const float* ac3_w2 = (const float*)d_in[9];
    const float* mid_w  = (const float*)d_in[10];
    const float* mid_b  = (const float*)d_in[11];
    const float* out_w  = (const float*)d_in[12];
    const float* out_b  = (const float*)d_in[13];
    float* out = (float*)d_out;

    const size_t N = (size_t)NVOX;
    float* ws   = (float*)d_ws;
    float* t8   = ws;            // 8N   weight-gen hidden layer
    float* wadp = ws + 8 * N;    // 27N  normalized adaptive weights
    float* bufA = ws + 35 * N;   // 8N
    float* bufB = ws + 43 * N;   // 8N
    float* f1   = ws + 51 * N;   // 1N
    float* f2   = ws + 52 * N;   // 1N
    unsigned short* Apk = (unsigned short*)(ws + 53 * N);  // packed A (bf16)

    dim3 cg(8, 32, 64);          // (W/16, H/4, D/2) tiles
    dim3 cb(256);
    dim3 ag(NVOX / 256);
    dim3 ab(256);
    const int PK1 = (1 * 7 * 32 * 16 + 255) / 256;   // pack grid, 1 M-tile
    const int PK2 = (2 * 7 * 32 * 16 + 255) / 256;   // pack grid, 2 M-tiles

    // ---- adaptive block 1 (weights from x) ----
    pack_wmma_A<1, 8><<<PK1, 256, 0, stream>>>(ac1_w1, Apk);
    conv3x3x3_wmma<1, MODE_RELU><<<cg, cb, 0, stream>>>(x, Apk, ac1_b1, t8);
    pack_wmma_A<2, 27><<<PK2, 256, 0, stream>>>(ac1_w2, Apk);
    conv3x3x3_wmma<2, MODE_NORM><<<cg, cb, 0, stream>>>(t8, Apk, nullptr, wadp);
    adaptive_conv_k<8><<<ag, ab, 0, stream>>>(x,    wadp, bufA);
    adaptive_conv_k<8><<<ag, ab, 0, stream>>>(bufA, wadp, bufB);
    adaptive_conv_k<8><<<ag, ab, 0, stream>>>(bufB, wadp, bufA);

    // ---- mid conv ----
    pack_wmma_A<1, 8><<<PK1, 256, 0, stream>>>(mid_w, Apk);
    conv3x3x3_wmma<1, MODE_BIAS><<<cg, cb, 0, stream>>>(bufA, Apk, mid_b, bufB);

    // ---- adaptive block 2 (weights from mid) ----
    pack_wmma_A<1, 8><<<PK1, 256, 0, stream>>>(ac2_w1, Apk);
    conv3x3x3_wmma<1, MODE_RELU><<<cg, cb, 0, stream>>>(bufB, Apk, ac2_b1, t8);
    pack_wmma_A<2, 27><<<PK2, 256, 0, stream>>>(ac2_w2, Apk);
    conv3x3x3_wmma<2, MODE_NORM><<<cg, cb, 0, stream>>>(t8, Apk, nullptr, wadp);
    adaptive_conv_k<8><<<ag, ab, 0, stream>>>(bufB, wadp, t8);
    adaptive_conv_k<8><<<ag, ab, 0, stream>>>(t8,   wadp, bufA);
    adaptive_conv_k<8><<<ag, ab, 0, stream>>>(bufA, wadp, bufB);   // bufB = mid2

    // ---- out conv (8 -> 1) ----
    pack_wmma_A<1, 1><<<PK1, 256, 0, stream>>>(out_w, Apk);
    conv3x3x3_wmma<1, MODE_OC1><<<cg, cb, 0, stream>>>(bufB, Apk, out_b, f1);

    // ---- adaptive block 3 (weights from mid2, applied to 1-ch final) ----
    pack_wmma_A<1, 8><<<PK1, 256, 0, stream>>>(ac3_w1, Apk);
    conv3x3x3_wmma<1, MODE_RELU><<<cg, cb, 0, stream>>>(bufB, Apk, ac3_b1, t8);
    pack_wmma_A<2, 27><<<PK2, 256, 0, stream>>>(ac3_w2, Apk);
    conv3x3x3_wmma<2, MODE_NORM><<<cg, cb, 0, stream>>>(t8, Apk, nullptr, wadp);
    adaptive_conv_k<1><<<ag, ab, 0, stream>>>(f1, wadp, f2);
    adaptive_conv_k<1><<<ag, ab, 0, stream>>>(f2, wadp, f1);
    adaptive_conv_k<1><<<ag, ab, 0, stream>>>(f1, wadp, out);
}
